// MoSELayer_78941498900674
// MI455X (gfx1250) — matrix operand order repaired and checked by reading, not compiled, not run on hardware
//
#include <hip/hip_runtime.h>
#include <hip/hip_bf16.h>

typedef _Float16 v16h __attribute__((ext_vector_type(16)));
typedef float    v8f  __attribute__((ext_vector_type(8)));
typedef unsigned int u32x4 __attribute__((ext_vector_type(4)));
typedef int          i32x4 __attribute__((ext_vector_type(4)));
typedef int          i32x8 __attribute__((ext_vector_type(8)));

#define CCH  768      // channels
#define NTOK 8192     // tokens = 8*32*32
#define TM   128
#define TN   128
#define TKK  32
#define NSTEP (CCH / TKK)   // 24

__device__ __forceinline__ float gelu_tanh(float x) {
  float x3 = x * x * x;
  return 0.5f * x * (1.0f + tanhf(0.7978845608028654f * (x + 0.044715f * x3)));
}

// --- Tensor Data Mover: DMA a 128x32 f16 tile (row stride CCH halfs) into LDS ---
// D# group0: count=1 | lds_addr | global_addr(57b) | type=2
// D# group1: data_size=1(2B) | tensor_dim0=768 | tensor_dim1=rows | tile_dim0=32 |
//            tile_dim1=128 | tensor_dim0_stride=768
__device__ __forceinline__ void tdm_load_tile_128x32(
    const _Float16* gsrc, unsigned ldsAddr, unsigned tensorDim1)
{
  unsigned long long ga = (unsigned long long)(uintptr_t)gsrc;
  u32x4 g0;
  g0[0] = 1u;                                       // count=1 (valid), no gather
  g0[1] = ldsAddr;                                  // LDS byte address
  g0[2] = (unsigned)ga;                             // global_addr[31:0]
  g0[3] = (unsigned)((ga >> 32) & 0x01FFFFFFu) | (2u << 30);  // addr[56:32] | type=2
  i32x8 g1;
  g1[0] = 0x00010000;                               // wg_mask=0, data_size=1 (2 bytes)
  g1[1] = (int)(768u << 16);                        // tensor_dim0[15:0]=768
  g1[2] = (int)(tensorDim1 << 16);                  // dim0[31:16]=0, tensor_dim1[15:0]
  g1[3] = (int)((tensorDim1 >> 16) | (32u << 16));  // tensor_dim1[31:16], tile_dim0=32
  g1[4] = 128;                                      // tile_dim1=128, tile_dim2=0
  g1[5] = 768;                                      // tensor_dim0_stride[31:0]=768
  g1[6] = 0;                                        // stride0[47:32]=0, stride1 lo=0
  g1[7] = 0;
  i32x4 z4 = {0, 0, 0, 0};
  i32x8 z8 = {0, 0, 0, 0, 0, 0, 0, 0};
  __builtin_amdgcn_tensor_load_to_lds(g0, g1, z4, z4, z8, 0);
}

// A fragment: 16x32 f16, A-tile row-major [M][TKK] in LDS.
// lane L: M = L%16 ; VGPR0-3 hold K = klo..klo+7, VGPR4-7 hold K = 16+klo..16+klo+7, klo = 8*(L>=16)
__device__ __forceinline__ v16h frag_a(const _Float16* sA, int mBase, int lane) {
  int m   = mBase + (lane & 15);
  int klo = (lane >> 4) << 3;
  const _Float16* p = sA + m * TKK + klo;
  v16h a;
  ((float4*)&a)[0] = *(const float4*)(p);        // K = klo .. klo+7
  ((float4*)&a)[1] = *(const float4*)(p + 16);   // K = 16+klo .. 16+klo+7
  return a;
}

// B fragment: 32x16 f16, B-tile stored out-major [N][TKK] in LDS.
// lane L: N = L%16 ; VGPR0-3 hold K = khi..khi+7, VGPR4-7 hold K = 16+khi..16+khi+7, khi = 8*(L>=16)
__device__ __forceinline__ v16h frag_b(const _Float16* sB, int nBase, int lane) {
  int n   = nBase + (lane & 15);
  int khi = (lane >> 4) << 3;
  const _Float16* p = sB + n * TKK + khi;
  v16h b;
  ((float4*)&b)[0] = *(const float4*)(p);        // K = khi .. khi+7
  ((float4*)&b)[1] = *(const float4*)(p + 16);   // K = 16+khi .. 16+khi+7
  return b;
}

// MODE 0: outH = gelu(A@B^T + bias) as f16   (gate layer1 / expert layer1)
// MODE 1: routed expert epilogue: if top_idx[m]==expert:
//         out32[bchw(m,col)] = (acc+bias)*gate_val[m] + s32[bchw(m,col)]
template <int MODE>
__global__ __launch_bounds__(256) void moe_gemm(
    const _Float16* __restrict__ A,    // [NTOK][CCH] f16
    const _Float16* __restrict__ BT,   // [CCH][CCH] f16, out-major
    const float*    __restrict__ bias, // [CCH]
    _Float16*       __restrict__ outH,
    float*          __restrict__ out32,
    const float*    __restrict__ res32,
    const int*      __restrict__ topIdx,
    const float*    __restrict__ gateVal,
    int expert)
{
  __shared__ _Float16 sA[2][TM * TKK];
  __shared__ _Float16 sB[2][TN * TKK];

  const int lane  = threadIdx.x & 31;
  const int wid   = threadIdx.x >> 5;     // 8 waves
  const int wm    = (wid & 1) * 64;       // wave M offset within tile
  const int wn    = (wid >> 1) * 32;      // wave N offset within tile
  const int tileM = blockIdx.x * TM;
  const int tileN = blockIdx.y * TN;

  v8f acc[4][2];
  #pragma unroll
  for (int j = 0; j < 2; ++j) {
    float bj = bias[tileN + wn + j * 16 + (lane & 15)];
    #pragma unroll
    for (int i = 0; i < 4; ++i)
      #pragma unroll
      for (int r = 0; r < 8; ++r) acc[i][j][r] = bj;
  }

  // prologue: wave 0 DMAs the first K-step tiles into buffer 0
  if (wid == 0) {
    tdm_load_tile_128x32(A  + (size_t)tileM * CCH, (unsigned)(uintptr_t)&sA[0][0], NTOK);
    tdm_load_tile_128x32(BT + (size_t)tileN * CCH, (unsigned)(uintptr_t)&sB[0][0], CCH);
  }

  for (int t = 0; t < NSTEP; ++t) {
    __builtin_amdgcn_s_wait_tensorcnt(0);   // issuing wave drains TDM; others pass free
    __syncthreads();                        // LDS tile ready; prev-step reads retired

    if (wid == 0 && t + 1 < NSTEP) {        // prefetch next K-step into other buffer
      int k0n = (t + 1) * TKK;
      tdm_load_tile_128x32(A  + (size_t)tileM * CCH + k0n,
                           (unsigned)(uintptr_t)&sA[(t + 1) & 1][0], NTOK);
      tdm_load_tile_128x32(BT + (size_t)tileN * CCH + k0n,
                           (unsigned)(uintptr_t)&sB[(t + 1) & 1][0], CCH);
    }

    const _Float16* cA = sA[t & 1];
    const _Float16* cB = sB[t & 1];
    v16h af[4], bf[2];
    #pragma unroll
    for (int i = 0; i < 4; ++i) af[i] = frag_a(cA, wm + i * 16, lane);
    #pragma unroll
    for (int j = 0; j < 2; ++j) bf[j] = frag_b(cB, wn + j * 16, lane);

    #pragma unroll
    for (int i = 0; i < 4; ++i)
      #pragma unroll
      for (int j = 0; j < 2; ++j)
        acc[i][j] = __builtin_amdgcn_wmma_f32_16x16x32_f16(
            false, af[i], false, bf[j], (short)0, acc[i][j], false, false);
  }

  // C/D layout: VGPR r -> row = r + 8*(lane>=16), col = lane%16
  #pragma unroll
  for (int i = 0; i < 4; ++i) {
    int rowBase = tileM + wm + i * 16 + ((lane >> 4) << 3);
    #pragma unroll
    for (int j = 0; j < 2; ++j) {
      int col = tileN + wn + j * 16 + (lane & 15);
      #pragma unroll
      for (int r = 0; r < 8; ++r) {
        int m = rowBase + r;
        float v = acc[i][j][r];
        if (MODE == 0) {
          outH[(size_t)m * CCH + col] = (_Float16)gelu_tanh(v);
        } else {
          if (topIdx[m] == expert) {
            float gv = gateVal[m];
            int b = m >> 10, hw = m & 1023;
            size_t idx = ((size_t)b * CCH + col) * 1024 + hw;
            out32[idx] = v * gv + res32[idx];
          }
        }
      }
    }
  }
}

// Gate layer 2 + softmax + top-1: one wave per token.
__global__ __launch_bounds__(256) void gate2_kernel(
    const _Float16* __restrict__ H1,   // [NTOK][CCH] f16
    const float*    __restrict__ w2,   // [CCH][4]
    const float*    __restrict__ b2,   // [4]
    int*   __restrict__ topIdx,
    float* __restrict__ gateVal)
{
  const int lane = threadIdx.x & 31;
  const int wid  = threadIdx.x >> 5;
  const int n    = blockIdx.x * 8 + wid;
  float p0 = 0.f, p1 = 0.f, p2 = 0.f, p3 = 0.f;
  for (int c = lane; c < CCH; c += 32) {
    float h = (float)H1[(size_t)n * CCH + c];
    float4 w = *(const float4*)(w2 + c * 4);
    p0 += h * w.x; p1 += h * w.y; p2 += h * w.z; p3 += h * w.w;
  }
  #pragma unroll
  for (int off = 16; off > 0; off >>= 1) {
    p0 += __shfl_xor(p0, off, 32);
    p1 += __shfl_xor(p1, off, 32);
    p2 += __shfl_xor(p2, off, 32);
    p3 += __shfl_xor(p3, off, 32);
  }
  if (lane == 0) {
    float l[4] = {p0 + b2[0], p1 + b2[1], p2 + b2[2], p3 + b2[3]};
    float mx = l[0]; int ti = 0;
    #pragma unroll
    for (int e = 1; e < 4; ++e) if (l[e] > mx) { mx = l[e]; ti = e; }
    float s = 0.f;
    #pragma unroll
    for (int e = 0; e < 4; ++e) s += expf(l[e] - mx);
    topIdx[n]  = ti;
    gateVal[n] = 1.0f / s;   // softmax prob of the argmax
  }
}

// Transpose+convert one 768x768 f32 [in][out] matrix -> f16 [out][in]; blockIdx.z picks the matrix.
__global__ __launch_bounds__(256) void wconvert_kernel(
    const float* __restrict__ gw1, const float* __restrict__ ew1,
    const float* __restrict__ ew2, _Float16* __restrict__ wt)
{
  __shared__ float tile[32][33];
  int mat = blockIdx.z;
  const float* src = (mat == 0) ? gw1
                   : (mat < 5)  ? ew1 + (size_t)(mat - 1) * CCH * CCH
                                : ew2 + (size_t)(mat - 5) * CCH * CCH;
  _Float16* dst = wt + (size_t)mat * CCH * CCH;
  int tx = threadIdx.x, ty = threadIdx.y;
  #pragma unroll
  for (int k = 0; k < 4; ++k) {
    int in  = blockIdx.y * 32 + ty + k * 8;
    int out = blockIdx.x * 32 + tx;
    tile[ty + k * 8][tx] = src[(size_t)in * CCH + out];
  }
  __syncthreads();
  #pragma unroll
  for (int k = 0; k < 4; ++k) {
    int out = blockIdx.x * 32 + ty + k * 8;
    int in  = blockIdx.y * 32 + tx;
    dst[(size_t)out * CCH + in] = (_Float16)tile[tx][ty + k * 8];
  }
}

// s32 BCHW f32 -> token-major f16 X[N][C] via LDS transpose. grid (32,24,8), block (32,8)
__global__ __launch_bounds__(256) void xconvert_kernel(
    const float* __restrict__ s32, _Float16* __restrict__ Xh)
{
  __shared__ float tile[32][33];
  int b = blockIdx.z;
  int tx = threadIdx.x, ty = threadIdx.y;
  #pragma unroll
  for (int k = 0; k < 4; ++k) {
    int c  = blockIdx.y * 32 + ty + k * 8;
    int hw = blockIdx.x * 32 + tx;
    tile[ty + k * 8][tx] = s32[((size_t)b * CCH + c) * 1024 + hw];
  }
  __syncthreads();
  #pragma unroll
  for (int k = 0; k < 4; ++k) {
    int hw = blockIdx.x * 32 + ty + k * 8;
    int c  = blockIdx.y * 32 + tx;
    Xh[((size_t)b * 1024 + hw) * CCH + c] = (_Float16)tile[tx][ty + k * 8];
  }
}

__global__ void copy4_kernel(const float4* __restrict__ s, float4* __restrict__ d, int n4) {
  int i = blockIdx.x * blockDim.x + threadIdx.x;
  if (i < n4) d[i] = s[i];
}

extern "C" void kernel_launch(void* const* d_in, const int* in_sizes, int n_in,
                              void* d_out, int out_size, void* d_ws, size_t ws_size,
                              hipStream_t stream) {
  const float* s4  = (const float*)d_in[0];
  const float* s8  = (const float*)d_in[1];
  const float* s16 = (const float*)d_in[2];
  const float* s32 = (const float*)d_in[3];
  const float* gw1 = (const float*)d_in[4];
  const float* gb1 = (const float*)d_in[5];
  const float* gw2 = (const float*)d_in[6];
  const float* gb2 = (const float*)d_in[7];
  const float* ew1 = (const float*)d_in[8];
  const float* eb1 = (const float*)d_in[9];
  const float* ew2 = (const float*)d_in[10];
  const float* eb2 = (const float*)d_in[11];

  float* out = (float*)d_out;
  const size_t n_s4 = 8ull * 96 * 256 * 256;
  const size_t n_s8 = 8ull * 192 * 128 * 128;
  const size_t n_s16 = 8ull * 384 * 64 * 64;
  float* out32 = out + n_s4 + n_s8 + n_s16;

  // workspace layout
  _Float16* Xh = (_Float16*)d_ws;                 // [NTOK][CCH]
  _Float16* WT = Xh + (size_t)NTOK * CCH;         // 9 x [CCH][CCH] out-major f16
  _Float16* H1 = WT + 9ull * CCH * CCH;           // gate hidden
  _Float16* H2 = H1 + (size_t)NTOK * CCH;         // expert hidden (reused)
  int*   topIdx  = (int*)(H2 + (size_t)NTOK * CCH);
  float* gateVal = (float*)(topIdx + NTOK);

  // pass-through copies
  copy4_kernel<<<(int)(n_s4 / 4 + 255) / 256, 256, 0, stream>>>(
      (const float4*)s4, (float4*)out, (int)(n_s4 / 4));
  copy4_kernel<<<(int)(n_s8 / 4 + 255) / 256, 256, 0, stream>>>(
      (const float4*)s8, (float4*)(out + n_s4), (int)(n_s8 / 4));
  copy4_kernel<<<(int)(n_s16 / 4 + 255) / 256, 256, 0, stream>>>(
      (const float4*)s16, (float4*)(out + n_s4 + n_s8), (int)(n_s16 / 4));

  // f16 conversions
  xconvert_kernel<<<dim3(32, 24, 8), dim3(32, 8), 0, stream>>>(s32, Xh);
  wconvert_kernel<<<dim3(24, 24, 9), dim3(32, 8), 0, stream>>>(gw1, ew1, ew2, WT);

  dim3 gg(NTOK / TM, CCH / TN);  // (64, 6)

  // gate layer 1 (WMMA) then small gate head + routing
  moe_gemm<0><<<gg, 256, 0, stream>>>(Xh, WT, gb1, H1, nullptr, nullptr, nullptr, nullptr, 0);
  gate2_kernel<<<NTOK / 8, 256, 0, stream>>>(H1, gw2, gb2, topIdx, gateVal);

  // experts: dense compute, routed epilogue writes out32 (each token exactly once)
  for (int e = 0; e < 4; ++e) {
    moe_gemm<0><<<gg, 256, 0, stream>>>(Xh, WT + (size_t)(1 + e) * CCH * CCH,
                                        eb1 + e * CCH, H2, nullptr, nullptr, nullptr, nullptr, 0);
    moe_gemm<1><<<gg, 256, 0, stream>>>(H2, WT + (size_t)(5 + e) * CCH * CCH,
                                        eb2 + e * CCH, nullptr, out32, s32, topIdx, gateVal, e);
  }
}
